// TransformerPairBiasLayer_22780506538104
// MI455X (gfx1250) — matrix operand order repaired and checked by reading, not compiled, not run on hardware
//
#include <hip/hip_runtime.h>
#include <hip/hip_bf16.h>

typedef __attribute__((ext_vector_type(16))) __bf16 v16bf;
typedef __attribute__((ext_vector_type(8)))  float  v8f;
typedef __attribute__((ext_vector_type(4)))  unsigned int v4u;
typedef __attribute__((ext_vector_type(8)))  int v8i;
typedef __attribute__((ext_vector_type(4)))  int v4i;
typedef __bf16 bf16;

#define BDIM 2
#define NSEQ 768
#define CS   1024
#define CZ   128
#define NH   16
#define HD   64
#define MTOK (BDIM * NSEQ)   // 1536

// GEMM blocking
#define NB   512             // block column width
#define KC   64              // K chunk
#define KCP  (KC + 8)        // padded LDS pitch for A tile (elements)
#define NBP  (NB + 8)        // padded LDS pitch for B tile (elements)
#define A_BUF_BYTES 4096     // 16*KCP*2 = 2304 used (+slack for TDM trailing pad)
#define B_BUF_BYTES 66816    // KC*NBP*2 = 66560 used (+slack)
#define GEMM_LDS (2 * A_BUF_BYTES + 2 * B_BUF_BYTES)

#if defined(__AMDGCN__) && __has_builtin(__builtin_amdgcn_tensor_load_to_lds)
#define HAVE_TDM 1
#else
#define HAVE_TDM 0
#endif

__device__ inline bf16 to_bf16(float x) { return (bf16)x; }

__device__ inline v8f wmma_bf16(v16bf a, v16bf b, v8f c) {
  // D = A(16x32) * B(32x16) + C(16x16 f32)
  return __builtin_amdgcn_wmma_f32_16x16x32_bf16(false, a, false, b, (short)0, c, false, false);
}

// A fragment 16x32 (MxK), row-major src, leading dim lda (elements).
__device__ inline v16bf load_a_frag(const bf16* A, int lda, int m0, int k0) {
  int lane = threadIdx.x & 31;
  const bf16* p = A + (long)(m0 + (lane & 15)) * lda + k0 + ((lane >> 4) << 3);
  v16bf a;
#pragma unroll
  for (int e = 0; e < 8; ++e) a[e] = p[e];
#pragma unroll
  for (int e = 0; e < 8; ++e) a[e + 8] = p[e + 16];
  return a;
}

// B fragment 32x16 (KxN), row-major src: lane%16 = N, elements = K (table order).
__device__ inline v16bf load_b_frag(const bf16* B, int ldb, int k0, int n0) {
  int lane = threadIdx.x & 31;
  const bf16* p = B + (long)(k0 + ((lane >> 4) << 3)) * ldb + n0 + (lane & 15);
  v16bf b;
#pragma unroll
  for (int e = 0; e < 8; ++e) b[e] = p[(long)e * ldb];
#pragma unroll
  for (int e = 0; e < 8; ++e) b[e + 8] = p[(long)(e + 16) * ldb];
  return b;
}

// C layout: VGPR v, lanes 0-15 -> (M=v, N=lane); lanes 16-31 -> (M=v+8, N=lane-16)
__device__ inline void c_coords(int& rb, int& col) {
  int lane = threadIdx.x & 31;
  col = lane & 15;
  rb  = (lane >> 4) << 3;
}

// ------------- TDM: 2D tile (tile_w x tile_h elements, 2B each) -> LDS -------------
#if HAVE_TDM
__device__ inline void tdm_issue_2d(unsigned lds_off, const bf16* gaddr,
                                    int tile_w, int tile_h, long row_stride_elems,
                                    int pad_interval, int pad_amount) {
  unsigned long long ga = (unsigned long long)gaddr;
  v4u g0;
  g0[0] = 1u;                                                  // count = 1 (valid)
  g0[1] = lds_off;                                             // lds_addr
  g0[2] = (unsigned)(ga & 0xFFFFFFFFu);                        // global_addr[31:0]
  g0[3] = (unsigned)((ga >> 32) & 0x01FFFFFFu) | (2u << 30);   // global_addr[56:32] | type=2
  unsigned td0 = (unsigned)tile_w;    // tensor_dim0 == tile width -> no OOB clamp
  unsigned td1 = (unsigned)tile_h;
  unsigned long long st0 = (unsigned long long)row_stride_elems;
  v8i g1;
  g1[0] = (int)(0x00010000u | (1u << 20) |                     // data_size=2B | pad_enable
                ((unsigned)pad_interval << 22) |
                ((unsigned)pad_amount << 25));
  g1[1] = (int)((td0 & 0xFFFFu) << 16);                        // tensor_dim0 lo
  g1[2] = (int)((td0 >> 16) | ((td1 & 0xFFFFu) << 16));        // tensor_dim0 hi | dim1 lo
  g1[3] = (int)((td1 >> 16) | ((unsigned)(tile_w & 0xFFFF) << 16)); // dim1 hi | tile_dim0
  g1[4] = (int)((unsigned)tile_h & 0xFFFFu);                   // tile_dim1 (tile_dim2 = 0)
  g1[5] = (int)(st0 & 0xFFFFFFFFu);                            // tensor_dim0_stride lo
  g1[6] = (int)((st0 >> 32) & 0xFFFFu);                        // stride hi (dim1_stride = 0)
  g1[7] = 0;
  v4i z4 = {0, 0, 0, 0};
#if __clang_major__ >= 23
  v8i z8 = {0, 0, 0, 0, 0, 0, 0, 0};
  __builtin_amdgcn_tensor_load_to_lds(g0, g1, z4, z4, z8, 0);
#else
  __builtin_amdgcn_tensor_load_to_lds(g0, g1, z4, z4, 0);
#endif
}
#endif

__device__ inline unsigned lds_off_of(const void* p) {
  return (unsigned)(unsigned long long)p;   // low 32 bits of generic addr = LDS offset
}

// cooperative fallback staging (mirrors TDM padded pitches)
__device__ inline void stage_coop(const bf16* A, int lda, const bf16* Bm, int ldb,
                                  int m0, int n0blk, int c, bf16* dA, bf16* dB) {
  for (int i = threadIdx.x; i < (16 * KC) / 4; i += 256) {
    int r = (i * 4) / KC, x = (i * 4) % KC;
    *(unsigned long long*)(dA + r * KCP + x) =
        *(const unsigned long long*)(A + (long)(m0 + r) * lda + c * KC + x);
  }
  for (int i = threadIdx.x; i < (KC * NB) / 4; i += 256) {
    int r = (i * 4) / NB, x = (i * 4) % NB;
    *(unsigned long long*)(dB + r * NBP + x) =
        *(const unsigned long long*)(Bm + (long)(c * KC + r) * ldb + n0blk + x);
  }
}

// Preload the full chunk's fragments, then fire 8 back-to-back WMMAs
// (independent accumulators -> no D->A/B hazards, one dscnt wait total).
__device__ inline void chunk_compute(const bf16* cA, const bf16* cB, int wave, v8f acc[4]) {
  int n0 = wave * 64;
  v16bf a0 = load_a_frag(cA, KCP, 0, 0);
  v16bf a1 = load_a_frag(cA, KCP, 0, 32);
  v16bf b[8];
#pragma unroll
  for (int t = 0; t < 4; ++t) {
    b[t]     = load_b_frag(cB, NBP, 0,  n0 + t * 16);
    b[4 + t] = load_b_frag(cB, NBP, 32, n0 + t * 16);
  }
#pragma unroll
  for (int t = 0; t < 4; ++t) acc[t] = wmma_bf16(a0, b[t], acc[t]);
#pragma unroll
  for (int t = 0; t < 4; ++t) acc[t] = wmma_bf16(a1, b[4 + t], acc[t]);
}

// TDM-staged, double-buffered block GEMM core. acc[4] = this wave's 16x64 tile.
__device__ inline void gemm_block_stage(const bf16* A, int lda, const bf16* Bm, int ldb,
                                        int m0, int n0blk, int K, v8f acc[4], char* smem) {
  int wave = threadIdx.x >> 5;
  int nchunks = K / KC;
  char* aBase = smem;
  char* bBase = smem + 2 * A_BUF_BYTES;
#if HAVE_TDM
  // TDM pads: A row = 64 elems = 32 DWORDs -> interval code 4; B row = 512 elems = 256 DWORDs
  // -> interval code 7. pad_amount code 3 = 4 DWORDs (16B) per row.
  if (wave == 0) {
    tdm_issue_2d(lds_off_of(aBase), A + (long)m0 * lda, KC, 16, lda, 4, 3);
    tdm_issue_2d(lds_off_of(bBase), Bm + n0blk, NB, KC, ldb, 7, 3);
    __builtin_amdgcn_s_wait_tensorcnt(0);
  }
  __syncthreads();
  for (int c = 0; c < nchunks; ++c) {
    int cur = c & 1, nxt = cur ^ 1;
    const bf16* cA = (const bf16*)(aBase + cur * A_BUF_BYTES);
    const bf16* cB = (const bf16*)(bBase + cur * B_BUF_BYTES);
    bool more = (c + 1 < nchunks);
    if (wave == 0 && more) {
      tdm_issue_2d(lds_off_of(aBase + nxt * A_BUF_BYTES),
                   A + (long)m0 * lda + (c + 1) * KC, KC, 16, lda, 4, 3);
      tdm_issue_2d(lds_off_of(bBase + nxt * B_BUF_BYTES),
                   Bm + (long)(c + 1) * KC * ldb + n0blk, NB, KC, ldb, 7, 3);
    }
    chunk_compute(cA, cB, wave, acc);
    if (wave == 0 && more) __builtin_amdgcn_s_wait_tensorcnt(0);
    __syncthreads();
  }
#else
  for (int c = 0; c < nchunks; ++c) {
    bf16* cA = (bf16*)(aBase + (c & 1) * A_BUF_BYTES);
    bf16* cB = (bf16*)(bBase + (c & 1) * B_BUF_BYTES);
    stage_coop(A, lda, Bm, ldb, m0, n0blk, c, cA, cB);
    __syncthreads();
    chunk_compute(cA, cB, wave, acc);
    __syncthreads();
  }
#endif
}

// ---------------- weight conversion ----------------
__global__ void k_pack_qkvg(const float* Wq, const float* Wkv, const float* Wg, bf16* dst) {
  long idx = (long)blockIdx.x * 256 + threadIdx.x;
  if (idx >= (long)CS * 4096) return;
  int k = (int)(idx >> 12);
  int n = (int)(idx & 4095);
  float v;
  if (n < 1024)       v = Wq[(long)k * 1024 + n];
  else if (n < 3072)  v = Wkv[(long)k * 2048 + (n - 1024)];
  else                v = Wg[(long)k * 1024 + (n - 3072)];
  dst[idx] = to_bf16(v);
}

__global__ void k_cvt_bf16(const float* s, bf16* d, long n) {
  long i = (long)blockIdx.x * 256 + threadIdx.x;
  if (i < n) d[i] = to_bf16(s[i]);
}

// ---------------- layernorm over CS, bf16 out ----------------
__global__ void k_ln_rows(const float* src, const float* g, const float* bb, bf16* dst) {
  __shared__ float red[256];
  long row = blockIdx.x;
  const float* x = src + row * CS;
  float s = 0.f, s2 = 0.f;
  for (int c = threadIdx.x; c < CS; c += 256) { float v = x[c]; s += v; s2 += v * v; }
  red[threadIdx.x] = s; __syncthreads();
  for (int o = 128; o; o >>= 1) { if (threadIdx.x < o) red[threadIdx.x] += red[threadIdx.x + o]; __syncthreads(); }
  float mean = red[0] * (1.f / CS); __syncthreads();
  red[threadIdx.x] = s2; __syncthreads();
  for (int o = 128; o; o >>= 1) { if (threadIdx.x < o) red[threadIdx.x] += red[threadIdx.x + o]; __syncthreads(); }
  float var  = red[0] * (1.f / CS) - mean * mean;
  float rstd = rsqrtf(var + 1e-5f);
  for (int c = threadIdx.x; c < CS; c += 256)
    dst[row * CS + c] = to_bf16((x[c] - mean) * rstd * g[c] + bb[c]);
}

// ---------------- GEMM kernels (TDM-staged core + epilogues) ----------------
__global__ void __launch_bounds__(256)
k_gemm_plain(const bf16* A, const bf16* B, float* C, int N, int K) {
  extern __shared__ char smem[];
  int m0 = blockIdx.x * 16, n0b = blockIdx.y * NB;
  v8f acc[4] = {};
  gemm_block_stage(A, K, B, N, m0, n0b, K, acc, smem);
  int wave = threadIdx.x >> 5, rb, col; c_coords(rb, col);
  int n0 = n0b + wave * 64;
#pragma unroll
  for (int t = 0; t < 4; ++t)
#pragma unroll
    for (int v = 0; v < 8; ++v)
      C[(long)(m0 + rb + v) * N + n0 + t * 16 + col] = acc[t][v];
}

__global__ void __launch_bounds__(256)
k_gemm_resid(const bf16* A, const bf16* B, const float* X, float* C, int N, int K) {
  extern __shared__ char smem[];
  int m0 = blockIdx.x * 16, n0b = blockIdx.y * NB;
  v8f acc[4] = {};
  gemm_block_stage(A, K, B, N, m0, n0b, K, acc, smem);
  int wave = threadIdx.x >> 5, rb, col; c_coords(rb, col);
  int n0 = n0b + wave * 64;
#pragma unroll
  for (int t = 0; t < 4; ++t)
#pragma unroll
    for (int v = 0; v < 8; ++v) {
      long o = (long)(m0 + rb + v) * N + n0 + t * 16 + col;
      C[o] = X[o] + acc[t][v];
    }
}

__global__ void __launch_bounds__(256)
k_gemm_bias_relu(const bf16* A, const bf16* B, const float* bias, bf16* D, int N, int K) {
  extern __shared__ char smem[];
  int m0 = blockIdx.x * 16, n0b = blockIdx.y * NB;
  v8f acc[4] = {};
  gemm_block_stage(A, K, B, N, m0, n0b, K, acc, smem);
  int wave = threadIdx.x >> 5, rb, col; c_coords(rb, col);
  int n0 = n0b + wave * 64;
#pragma unroll
  for (int t = 0; t < 4; ++t) {
    int c = n0 + t * 16 + col;
    float bv = bias[c];
#pragma unroll
    for (int v = 0; v < 8; ++v)
      D[(long)(m0 + rb + v) * N + c] = to_bf16(fmaxf(acc[t][v] + bv, 0.f));
  }
}

__global__ void __launch_bounds__(256)
k_gemm_ffn2(const bf16* A, const bf16* B, const float* bias, const float* X,
            const float* mask, float* Out, int N, int K) {
  extern __shared__ char smem[];
  int m0 = blockIdx.x * 16, n0b = blockIdx.y * NB;
  v8f acc[4] = {};
  gemm_block_stage(A, K, B, N, m0, n0b, K, acc, smem);
  int wave = threadIdx.x >> 5, rb, col; c_coords(rb, col);
  int n0 = n0b + wave * 64;
#pragma unroll
  for (int t = 0; t < 4; ++t) {
    int c = n0 + t * 16 + col;
    float bv = bias[c];
#pragma unroll
    for (int v = 0; v < 8; ++v) {
      int r = m0 + rb + v;
      long o = (long)r * N + c;
      Out[o] = X[o] + (acc[t][v] + bv) * mask[r];
    }
  }
}

// ---------------- repack q/k/v/gate from fused GEMM output ----------------
__global__ void k_repack(const float* qkvg, bf16* qb, bf16* kT, bf16* vb, float* sg) {
  long idx = (long)blockIdx.x * 256 + threadIdx.x;   // MTOK*CS
  if (idx >= (long)MTOK * CS) return;
  int r = (int)(idx >> 10);
  int c = (int)(idx & 1023);
  const float* row = qkvg + (long)r * 4096;
  qb[idx] = to_bf16(row[c]);
  vb[idx] = to_bf16(row[2048 + c]);
  sg[idx] = 1.f / (1.f + __expf(-row[3072 + c]));
  int b = r / NSEQ, i = r % NSEQ;
  int h = c >> 6, d = c & 63;
  kT[((long)(b * NH + h) * HD + d) * NSEQ + i] = to_bf16(row[1024 + c]);
}

// ---------------- pair bias: b = LN(z) @ Wb + mask, layout (B, i, j, H) ----------------
__global__ void __launch_bounds__(256)
k_bias_z(const float* z, const float* zg, const float* zb,
         const float* Wb, const float* mask, float* bout) {
  __shared__ float sWb[CZ * NH];
  __shared__ float sg_[CZ], sb_[CZ];
  for (int i = threadIdx.x; i < CZ * NH; i += 256) sWb[i] = Wb[i];
  for (int i = threadIdx.x; i < CZ; i += 256) { sg_[i] = zg[i]; sb_[i] = zb[i]; }
  __syncthreads();
  int wave = threadIdx.x >> 5, lane = threadIdx.x & 31;
  long row = (long)blockIdx.x * 8 + wave;
  const float* zr = z + row * CZ;
  int c0 = lane * 4;
  float4 zv = *(const float4*)(zr + c0);
  float s  = zv.x + zv.y + zv.z + zv.w;
  float s2 = zv.x * zv.x + zv.y * zv.y + zv.z * zv.z + zv.w * zv.w;
  for (int o = 16; o; o >>= 1) { s += __shfl_xor(s, o); s2 += __shfl_xor(s2, o); }
  float mu   = s * (1.f / CZ);
  float var  = s2 * (1.f / CZ) - mu * mu;
  float rstd = rsqrtf(var + 1e-5f);
  float zn[4];
  zn[0] = (zv.x - mu) * rstd * sg_[c0 + 0] + sb_[c0 + 0];
  zn[1] = (zv.y - mu) * rstd * sg_[c0 + 1] + sb_[c0 + 1];
  zn[2] = (zv.z - mu) * rstd * sg_[c0 + 2] + sb_[c0 + 2];
  zn[3] = (zv.w - mu) * rstd * sg_[c0 + 3] + sb_[c0 + 3];
  float acc[NH];
#pragma unroll
  for (int h = 0; h < NH; ++h) acc[h] = 0.f;
#pragma unroll
  for (int j = 0; j < 4; ++j) {
    const float* wrow = &sWb[(c0 + j) * NH];
#pragma unroll
    for (int h = 0; h < NH; ++h) acc[h] += zn[j] * wrow[h];
  }
  for (int o = 16; o; o >>= 1)
#pragma unroll
    for (int h = 0; h < NH; ++h) acc[h] += __shfl_xor(acc[h], o);
  if (lane == 0) {
    long t = row; int jj = (int)(t % NSEQ); t /= NSEQ; int ii = (int)(t % NSEQ); int bi = (int)(t / NSEQ);
    float mterm = 1e8f * (mask[bi * NSEQ + jj] - 1.f) + 1e8f * (mask[bi * NSEQ + ii] - 1.f);
    float* o = bout + (((long)bi * NSEQ + ii) * NSEQ + jj) * NH;
#pragma unroll
    for (int h = 0; h < NH; ++h) o[h] = acc[h] + mterm;
  }
}

// ---------------- attention: one block per (b, h, 16-row i-tile) ----------------
__global__ void __launch_bounds__(256)
k_attn(const bf16* qb, const bf16* kT, const bf16* vb, const float* bias, float* upd) {
  extern __shared__ char smem[];
  float* sS   = (float*)smem;                       // 16*768 f32   = 48KB
  bf16*  sP   = (bf16*)(smem + 16 * NSEQ * 4);      // 16*768 bf16  = 24KB
  float* sRow = (float*)(smem + 16 * NSEQ * 4 + 16 * NSEQ * 2);
  int it = blockIdx.x, h = blockIdx.y, bi = blockIdx.z;
  int wave = threadIdx.x >> 5, lane = threadIdx.x & 31;
  int i0 = it * 16;
  int rb, col; c_coords(rb, col);
  const bf16* Aq = qb + (long)(bi * NSEQ + i0) * CS + h * HD;        // lda=CS
  const bf16* Bk = kT + (long)(bi * NH + h) * HD * NSEQ;             // ldb=NSEQ
  const float scale = 0.125f;                                        // 1/sqrt(64)
  // phase 1: S = scale * q kT + bias (q hoisted, K-tiles software-pipelined)
  v16bf aq0 = load_a_frag(Aq, CS, 0, 0);
  v16bf aq1 = load_a_frag(Aq, CS, 0, 32);
  v16bf b0 = load_b_frag(Bk, NSEQ, 0, wave * 16);
  v16bf b1 = load_b_frag(Bk, NSEQ, 32, wave * 16);
  for (int jt = wave; jt < NSEQ / 16; jt += 8) {
    v16bf nb0 = b0, nb1 = b1;
    if (jt + 8 < NSEQ / 16) {
      nb0 = load_b_frag(Bk, NSEQ, 0, (jt + 8) * 16);
      nb1 = load_b_frag(Bk, NSEQ, 32, (jt + 8) * 16);
    }
    v8f acc = {};
    acc = wmma_bf16(aq0, b0, acc);
    acc = wmma_bf16(aq1, b1, acc);
    int j0 = jt * 16 + col;
    const float* bp = bias + (((long)bi * NSEQ + (i0 + rb)) * NSEQ + j0) * NH + h;
#pragma unroll
    for (int v = 0; v < 8; ++v)
      sS[(rb + v) * NSEQ + j0] = acc[v] * scale + bp[(long)v * NSEQ * NH];
    b0 = nb0; b1 = nb1;
  }
  __syncthreads();
  // phase 2: softmax rows (unnormalized exp -> bf16, rowsum kept)
  for (int r = wave; r < 16; r += 8) {
    float vals[NSEQ / 32];
    float mx = -3.0e38f;
#pragma unroll
    for (int t = 0; t < NSEQ / 32; ++t) { vals[t] = sS[r * NSEQ + lane + t * 32]; mx = fmaxf(mx, vals[t]); }
    for (int o = 16; o; o >>= 1) mx = fmaxf(mx, __shfl_xor(mx, o));
    float sum = 0.f;
#pragma unroll
    for (int t = 0; t < NSEQ / 32; ++t) {
      float e = __expf(vals[t] - mx);
      sum += e;
      sP[r * NSEQ + lane + t * 32] = to_bf16(e);
    }
    for (int o = 16; o; o >>= 1) sum += __shfl_xor(sum, o);
    if (lane == 0) sRow[r] = sum;
  }
  __syncthreads();
  // phase 3: U = P V (software-pipelined), normalize by rowsum
  if (wave < 4) {
    const bf16* Bv = vb + (long)(bi * NSEQ) * CS + h * HD;           // ldb=CS, K=j
    v8f acc = {};
    v16bf a = load_a_frag(sP, NSEQ, 0, 0);
    v16bf b = load_b_frag(Bv, CS, 0, wave * 16);
    for (int k0 = 0; k0 < NSEQ; k0 += 32) {
      v16bf an = a, bn = b;
      if (k0 + 32 < NSEQ) {
        an = load_a_frag(sP, NSEQ, 0, k0 + 32);
        bn = load_b_frag(Bv, CS, k0 + 32, wave * 16);
      }
      acc = wmma_bf16(a, b, acc);
      a = an; b = bn;
    }
#pragma unroll
    for (int v = 0; v < 8; ++v) {
      int r = rb + v;
      upd[(long)(bi * NSEQ + i0 + r) * CS + h * HD + wave * 16 + col] = acc[v] / sRow[r];
    }
  }
}

__global__ void k_gate(const float* upd, const float* sg, bf16* gated) {
  long i = (long)blockIdx.x * 256 + threadIdx.x;
  if (i < (long)MTOK * CS) gated[i] = to_bf16(upd[i] * sg[i]);
}

extern "C" void kernel_launch(void* const* d_in, const int* in_sizes, int n_in,
                              void* d_out, int out_size, void* d_ws, size_t ws_size,
                              hipStream_t stream) {
  const float* x      = (const float*)d_in[0];
  const float* z      = (const float*)d_in[1];
  const float* x_mask = (const float*)d_in[2];
  const float* ln_s_g = (const float*)d_in[3];
  const float* ln_s_b = (const float*)d_in[4];
  const float* ln_z_g = (const float*)d_in[5];
  const float* ln_z_b = (const float*)d_in[6];
  const float* Wq     = (const float*)d_in[7];
  const float* Wkv    = (const float*)d_in[8];
  const float* Wb     = (const float*)d_in[9];
  const float* Wgate  = (const float*)d_in[10];
  const float* Wout   = (const float*)d_in[11];
  const float* ffg    = (const float*)d_in[12];
  const float* ffb    = (const float*)d_in[13];
  const float* W1     = (const float*)d_in[14];
  const float* b1     = (const float*)d_in[15];
  const float* W2     = (const float*)d_in[16];
  const float* b2     = (const float*)d_in[17];

  char* base = (char*)d_ws;
  size_t off = 0;
  auto alloc = [&](size_t bytes) -> void* {
    off = (off + 255) & ~(size_t)255;
    void* p = base + off;
    off += bytes;
    return p;
  };
  bf16*  Wqkvg  = (bf16*)alloc((size_t)CS * 4096 * 2);
  bf16*  Woutb  = (bf16*)alloc((size_t)CS * CS * 2);
  bf16*  W1b    = (bf16*)alloc((size_t)CS * 2048 * 2);
  bf16*  W2b    = (bf16*)alloc((size_t)2048 * CS * 2);
  bf16*  xln    = (bf16*)alloc((size_t)MTOK * CS * 2);
  float* qkvg   = (float*)alloc((size_t)MTOK * 4096 * 4);
  bf16*  q_b    = (bf16*)alloc((size_t)MTOK * CS * 2);
  bf16*  kT_b   = (bf16*)alloc((size_t)MTOK * CS * 2);
  bf16*  v_b    = (bf16*)alloc((size_t)MTOK * CS * 2);
  float* sgate  = (float*)alloc((size_t)MTOK * CS * 4);
  float* bval   = (float*)alloc((size_t)BDIM * NSEQ * NSEQ * NH * 4);
  float* updf   = (float*)alloc((size_t)MTOK * CS * 4);
  bf16*  gated  = (bf16*)alloc((size_t)MTOK * CS * 2);
  float* x1     = (float*)alloc((size_t)MTOK * CS * 4);
  bf16*  t_b    = (bf16*)alloc((size_t)MTOK * CS * 2);
  bf16*  h_b    = (bf16*)alloc((size_t)MTOK * 2048 * 2);

  // weight conversion to bf16
  k_pack_qkvg<<<(CS * 4096 + 255) / 256, 256, 0, stream>>>(Wq, Wkv, Wgate, Wqkvg);
  k_cvt_bf16<<<(CS * CS + 255) / 256, 256, 0, stream>>>(Wout, Woutb, (long)CS * CS);
  k_cvt_bf16<<<(CS * 2048 + 255) / 256, 256, 0, stream>>>(W1, W1b, (long)CS * 2048);
  k_cvt_bf16<<<(2048 * CS + 255) / 256, 256, 0, stream>>>(W2, W2b, (long)2048 * CS);

  // LN(x) -> bf16
  k_ln_rows<<<MTOK, 256, 0, stream>>>(x, ln_s_g, ln_s_b, xln);

  // fused q|kv|gate GEMM: (1536 x 1024) @ (1024 x 4096), TDM-staged
  k_gemm_plain<<<dim3(MTOK / 16, 4096 / NB), 256, GEMM_LDS, stream>>>(xln, Wqkvg, qkvg, 4096, CS);

  // repack to q bf16, kT (B,H,D,N), v bf16, sigmoid(gate)
  k_repack<<<(MTOK * CS + 255) / 256, 256, 0, stream>>>(qkvg, q_b, kT_b, v_b, sgate);

  // pair bias (streams all 604MB of z once)
  k_bias_z<<<(BDIM * NSEQ * NSEQ) / 8, 256, 0, stream>>>(z, ln_z_g, ln_z_b, Wb, x_mask, bval);

  // attention
  size_t attn_lds = (size_t)16 * NSEQ * 4 + (size_t)16 * NSEQ * 2 + 64;
  k_attn<<<dim3(NSEQ / 16, NH, BDIM), 256, attn_lds, stream>>>(q_b, kT_b, v_b, bval, updf);

  // gating
  k_gate<<<(MTOK * CS + 255) / 256, 256, 0, stream>>>(updf, sgate, gated);

  // out projection + residual: x1 = x + gated @ Wout
  k_gemm_resid<<<dim3(MTOK / 16, CS / NB), 256, GEMM_LDS, stream>>>(gated, Woutb, x, x1, CS, CS);

  // FFN
  k_ln_rows<<<MTOK, 256, 0, stream>>>(x1, ffg, ffb, t_b);
  k_gemm_bias_relu<<<dim3(MTOK / 16, 2048 / NB), 256, GEMM_LDS, stream>>>(t_b, W1b, b1, h_b, 2048, CS);
  k_gemm_ffn2<<<dim3(MTOK / 16, CS / NB), 256, GEMM_LDS, stream>>>(h_b, W2b, b2, x1, x_mask,
                                                                   (float*)d_out, CS, 2048);
}
